// MPNN_63745904607449
// MI455X (gfx1250) — compile-verified, matrix operand order checked by reading, not run on hardware
//
#include <hip/hip_runtime.h>
#include <hip/hip_bf16.h>

// ---------------------------------------------------------------------------
// MPNN (GCN stack) for MI455X / gfx1250, wave32 + bf16 WMMA + TDM staging.
// Dense H@W via v_wmma_f32_16x16x32_bf16 (f32 accumulate), W staged into LDS
// with tensor_load_to_lds (TENSORcnt), sparse aggregation via f32 global
// atomics (L2-resident destination), LN via wave32 shuffles.
// ---------------------------------------------------------------------------

enum { kDH = 128, kNGraphs = 128 };
#define LN_EPS 1e-5f
#define MT 4   // 16-row M tiles per block

typedef __attribute__((ext_vector_type(16))) __bf16 v16bf;
typedef __attribute__((ext_vector_type(8)))  float  v8f;
typedef __attribute__((ext_vector_type(4)))  unsigned v4u;
typedef __attribute__((ext_vector_type(8)))  int    v8i;
typedef __attribute__((ext_vector_type(4)))  int    v4i;

// Native f32 -> bf16 (RTNE); lowers to v_cvt_pk_bf16_f32 on gfx1250.
__device__ __forceinline__ __bf16 f2bf(float f) { return (__bf16)f; }

// --------------------------- dense GEMM (WMMA) -----------------------------
// Ht[M,128] = A[M,128] @ W[128,128]   (bf16 inputs, f32 accumulate)
// Agg[M,128] = bias + Ht * dinv^2    (fused GCN self-loop epilogue)
// Block = 256 threads = 8 waves; wave w owns N tile [16w,16w+16); the block
// covers MT=4 consecutive 16-row M tiles. W is staged once into LDS via the
// Tensor Data Mover, B fragments preloaded into registers, 16 WMMAs/wave.
__global__ __launch_bounds__(256)
void k_gemm128(const float* __restrict__ A, const float* __restrict__ W,
               float* __restrict__ Ht, float* __restrict__ Agg,
               const float* __restrict__ dinv, const float* __restrict__ bias,
               int M) {
  __shared__ float Wlds[kDH * kDH];   // 64 KB; sole LDS object -> offset 0

  const int tid  = threadIdx.x;
  const int wave = tid >> 5;
  const int lane = tid & 31;
  const int half = lane >> 4;     // lanes 0-15 vs 16-31
  const int mr   = lane & 15;     // row (A) / col (B) within tile
  const int kb   = half << 3;     // K sub-chunk base: 0 or 8
  const int n0   = wave << 4;
  const int m0   = blockIdx.x * (16 * MT);

  // ---- stage W (128x128 f32, 64KB) into LDS via Tensor Data Mover ----
#if __has_builtin(__builtin_amdgcn_tensor_load_to_lds)
  if (wave == 0) {
    unsigned long long ga = (unsigned long long)(const void*)W;
    v4u g0;
    g0[0] = 1u;                                   // count=1, user mode
    g0[1] = 0u;                                   // lds_addr = 0
    g0[2] = (unsigned)(ga & 0xFFFFFFFFu);         // global_addr[31:0]
    g0[3] = (unsigned)((ga >> 32) & 0x01FFFFFFu)  // global_addr[56:32]
          | (2u << 30);                           // type = 2 ("image")
    v8i g1;
    g1[0] = (int)(2u << 16);        // data_size=2 (4B), no multicast
    g1[1] = (int)(128u << 16);      // tensor_dim0[15:0]=128  (bits 79:48)
    g1[2] = (int)(128u << 16);      // tensor_dim1[15:0]=128  (bits 111:80)
    g1[3] = (int)(128u << 16);      // tile_dim0=128          (bits 127:112)
    g1[4] = 128;                    // tile_dim1=128, tile_dim2=0
    g1[5] = 128;                    // tensor_dim0_stride[31:0]=128
    g1[6] = 0;                      // stride hi / dim1_stride lo
    g1[7] = 0;
    v4i gz4 = {0, 0, 0, 0};
    v8i gz8 = {0, 0, 0, 0, 0, 0, 0, 0};
    // 6-arg flavor: (g0, g1, g2, g3, g4, cpol)
    __builtin_amdgcn_tensor_load_to_lds(g0, g1, gz4, gz4, gz8, 0);
    __builtin_amdgcn_s_wait_tensorcnt(0);
  }
#else
  for (int i = tid; i < (kDH * kDH) / 4; i += 256)
    ((float4*)Wlds)[i] = ((const float4*)W)[i];
#endif
  // The TDM writes LDS behind the compiler's back, and Wlds never escapes,
  // so a plain memory clobber is not enough (escape analysis still proves
  // Wlds unwritten). Launder the pointer through the asm: the compiler can
  // no longer reason about what it points to, so every load materializes.
  const float* wl = &Wlds[0];
  asm volatile("" : "+v"(wl) : : "memory");
  __syncthreads();

  // ---- preload all 4 B fragments (K chunks of 32) from LDS ----
  v16bf bfrag[4];
#pragma unroll
  for (int kt = 0; kt < 4; ++kt) {
    const int kB = kt * 32 + kb;
#pragma unroll
    for (int j = 0; j < 8; ++j)
      bfrag[kt][j]     = f2bf(wl[(kB + j) * kDH + n0 + mr]);
#pragma unroll
    for (int j = 0; j < 8; ++j)
      bfrag[kt][8 + j] = f2bf(wl[(kB + 16 + j) * kDH + n0 + mr]);
  }

  const float bn = bias[n0 + mr];

  // ---- MT M-tiles: load A fragment, 4 WMMAs, fused epilogue ----
  for (int mt = 0; mt < MT; ++mt) {
    const int mbase = m0 + mt * 16;
    if (mbase >= M) break;                  // uniform across block
    const int arow  = mbase + mr;
    const int arowc = arow < M ? arow : M - 1;
    if (mbase + 16 < M)                     // warm next tile's rows
      __builtin_prefetch(A + (size_t)(mbase + 16 + mr) * kDH, 0, 1);

    v8f c = {};
#pragma unroll
    for (int kt = 0; kt < 4; ++kt) {
      const float* pa = A + (size_t)arowc * kDH + kt * 32 + kb;
      v16bf a;
#pragma unroll
      for (int j = 0; j < 8; ++j) a[j]     = f2bf(pa[j]);
#pragma unroll
      for (int j = 0; j < 8; ++j) a[8 + j] = f2bf(pa[16 + j]);
      c = __builtin_amdgcn_wmma_f32_16x16x32_bf16(false, a, false, bfrag[kt],
                                                  (short)0, c, false, false);
    }

    // C/D layout: VGPR r -> row r + 8*half, col = mr. Fused self-loop+bias.
    const int rowbase = mbase + (half << 3);
    float* pht = Ht  + (size_t)rowbase * kDH + n0 + mr;
    float* pag = Agg + (size_t)rowbase * kDH + n0 + mr;
    const float* pdi = dinv + rowbase;
    if (mbase + 16 <= M) {                  // full tile: straight-line stores
#pragma unroll
      for (int r = 0; r < 8; ++r) {
        const float v  = c[r];
        const float di = pdi[r];
        pht[(size_t)r * kDH] = v;
        pag[(size_t)r * kDH] = bn + v * di * di;
      }
    } else {                                // partial final tile
#pragma unroll
      for (int r = 0; r < 8; ++r) {
        if (rowbase + r < M) {
          const float v  = c[r];
          const float di = pdi[r];
          pht[(size_t)r * kDH] = v;
          pag[(size_t)r * kDH] = bn + v * di * di;
        }
      }
    }
  }
}

// ----------------------------- small helpers -------------------------------
__global__ void k_fill1(float* p, int n) {
  int i = blockIdx.x * blockDim.x + threadIdx.x;
  if (i < n) p[i] = 1.0f;
}
__global__ void k_zero(float* p, int n) {
  int i = blockIdx.x * blockDim.x + threadIdx.x;
  if (i < n) p[i] = 0.0f;
}
__global__ void k_deg_count(const int* __restrict__ dst, float* deg, int E) {
  int e = blockIdx.x * blockDim.x + threadIdx.x;
  if (e < E) atomicAdd(&deg[dst[e]], 1.0f);
}
__global__ void k_rsqrt(float* p, int n) {
  int i = blockIdx.x * blockDim.x + threadIdx.x;
  if (i < n) p[i] = rsqrtf(p[i]);
}

// One wave32 per edge: gather 128-ch row of src, scatter-add into dst.
__global__ __launch_bounds__(256)
void k_edge_scatter(const float* __restrict__ Ht, const int* __restrict__ srcI,
                    const int* __restrict__ dstI, const float* __restrict__ dinv,
                    float* __restrict__ Agg, int E) {
  int e    = (blockIdx.x * blockDim.x + threadIdx.x) >> 5;
  int lane = threadIdx.x & 31;
  if (e >= E) return;
  int s = srcI[e], d = dstI[e];
  float w = dinv[s] * dinv[d];
  const float* hs = Ht + (size_t)s * kDH;
  float* ad = Agg + (size_t)d * kDH;
#pragma unroll
  for (int j = 0; j < 4; ++j) {
    int c = lane + (j << 5);
    atomicAdd(&ad[c], hs[c] * w);
  }
}

// h += relu(LN(agg) * gamma + beta); one wave32 per node.
__global__ __launch_bounds__(256)
void k_ln_relu_res(const float* __restrict__ Agg, const float* __restrict__ gamma,
                   const float* __restrict__ beta, float* __restrict__ H, int n) {
  int node = (blockIdx.x * blockDim.x + threadIdx.x) >> 5;
  int lane = threadIdx.x & 31;
  if (node >= n) return;
  const float* a = Agg + (size_t)node * kDH;
  float v[4], s = 0.f, sq = 0.f;
#pragma unroll
  for (int j = 0; j < 4; ++j) {
    v[j] = a[lane + (j << 5)];
    s += v[j]; sq += v[j] * v[j];
  }
#pragma unroll
  for (int off = 16; off > 0; off >>= 1) {
    s  += __shfl_xor(s,  off, 32);
    sq += __shfl_xor(sq, off, 32);
  }
  float mu  = s * (1.0f / kDH);
  float var = sq * (1.0f / kDH) - mu * mu;
  float ir  = rsqrtf(var + LN_EPS);
  float* h = H + (size_t)node * kDH;
#pragma unroll
  for (int j = 0; j < 4; ++j) {
    int c = lane + (j << 5);
    float t = (v[j] - mu) * ir * gamma[c] + beta[c];
    h[c] += fmaxf(t, 0.0f);
  }
}

// global_add_pool: one wave32 per node, atomic into [128,128] (L2-resident).
__global__ __launch_bounds__(256)
void k_pool(const float* __restrict__ H, const int* __restrict__ batch,
            float* __restrict__ Y, int n) {
  int node = (blockIdx.x * blockDim.x + threadIdx.x) >> 5;
  int lane = threadIdx.x & 31;
  if (node >= n) return;
  int g = batch[node];
  const float* h = H + (size_t)node * kDH;
  float* y = Y + (size_t)g * kDH;
#pragma unroll
  for (int j = 0; j < 4; ++j) {
    int c = lane + (j << 5);
    atomicAdd(&y[c], h[c]);
  }
}

// out[g] = relu(Y[g]@W1 + b1) @ W2 + b2; one 128-thread block per graph.
__global__ __launch_bounds__(128)
void k_readout(const float* __restrict__ Y, const float* __restrict__ W1,
               const float* __restrict__ b1, const float* __restrict__ W2,
               const float* __restrict__ b2, float* __restrict__ out) {
  __shared__ float red[kDH];
  int g = blockIdx.x, j = threadIdx.x;
  const float* y = Y + (size_t)g * kDH;
  float acc = b1[j];
  for (int k = 0; k < kDH; ++k) acc += y[k] * W1[(size_t)k * kDH + j];
  red[j] = fmaxf(acc, 0.0f) * W2[j];
  __syncthreads();
  for (int s = kDH >> 1; s > 0; s >>= 1) {
    if (j < s) red[j] += red[j + s];
    __syncthreads();
  }
  if (j == 0) out[g] = red[0] + b2[0];
}

// ---------------------------------------------------------------------------
extern "C" void kernel_launch(void* const* d_in, const int* in_sizes, int n_in,
                              void* d_out, int out_size, void* d_ws, size_t ws_size,
                              hipStream_t stream) {
  (void)n_in; (void)out_size; (void)ws_size;
  const float* x     = (const float*)d_in[0];
  const int*   ei    = (const int*)d_in[1];
  const int*   batch = (const int*)d_in[2];
  const float* W_enc = (const float*)d_in[3];
  const float* b_enc = (const float*)d_in[4];
  const float* W_blk = (const float*)d_in[5];
  const float* b_blk = (const float*)d_in[6];
  const float* gamma = (const float*)d_in[7];
  const float* beta  = (const float*)d_in[8];
  const float* W1    = (const float*)d_in[9];
  const float* b1    = (const float*)d_in[10];
  const float* W2    = (const float*)d_in[11];
  const float* b2    = (const float*)d_in[12];

  const int N = in_sizes[0] / kDH;   // 50000
  const int E = in_sizes[1] / 2;     // 800000
  const int* srcI = ei;              // edge_index[0]
  const int* dstI = ei + E;          // edge_index[1]

  // workspace layout
  size_t nodeF = (size_t)N * kDH;
  float* Ht   = (float*)d_ws;        // GEMM output (pre-aggregation h@W)
  float* bufA = Ht + nodeF;          // aggregation ping
  float* bufB = bufA + nodeF;        // aggregation pong / H
  float* dinv = bufB + nodeF;        // [N]
  float* Y    = dinv + N;            // [128,128]

  const int T = 256;
  int gN     = (N + T - 1) / T;
  int gEdge  = (int)(((size_t)E * 32 + T - 1) / T);
  int gWaveN = (int)(((size_t)N * 32 + T - 1) / T);
  int gGemm  = (N + 16 * MT - 1) / (16 * MT);

  // degrees (self-loop => start at 1), then dinv = rsqrt(deg) in place
  k_fill1<<<gN, T, 0, stream>>>(dinv, N);
  k_deg_count<<<(E + T - 1) / T, T, 0, stream>>>(dstI, dinv, E);
  k_rsqrt<<<gN, T, 0, stream>>>(dinv, N);

  auto gcn = [&](const float* Hin, const float* Wm, const float* bias, float* Agg) {
    k_gemm128<<<gGemm, T, 0, stream>>>(Hin, Wm, Ht, Agg, dinv, bias, N);
    k_edge_scatter<<<gEdge, T, 0, stream>>>(Ht, srcI, dstI, dinv, Agg, E);
  };

  // h = gcn(x, W_enc); h = gcn(h, W_blk[0])
  gcn(x, W_enc, b_enc, bufA);
  gcn(bufA, W_blk, b_blk, bufB);
  float* H = bufB;

  // residual DeepGCN blocks l = 0..2: h += relu(LN(gcn(h, W_blk[l])))
  for (int l = 0; l < 3; ++l) {
    gcn(H, W_blk + (size_t)l * kDH * kDH, b_blk + (size_t)l * kDH, bufA);
    k_ln_relu_res<<<gWaveN, T, 0, stream>>>(bufA, gamma + (size_t)l * kDH,
                                            beta + (size_t)l * kDH, H, N);
  }

  // pool + readout
  k_zero<<<(kNGraphs * kDH + T - 1) / T, T, 0, stream>>>(Y, kNGraphs * kDH);
  k_pool<<<gWaveN, T, 0, stream>>>(H, batch, Y, N);
  k_readout<<<kNGraphs, kDH, 0, stream>>>(Y, W1, b1, W2, b2, (float*)d_out);
}
